// Attention_77773267796265
// MI455X (gfx1250) — compile-verified, matrix operand order checked by reading, not run on hardware
//
#include <hip/hip_runtime.h>

// ---------------------------------------------------------------------------
// Types for CDNA5 WMMA (wave32): v_wmma_f32_16x16x32_bf16
// ---------------------------------------------------------------------------
typedef __attribute__((ext_vector_type(16))) __bf16          v16bf;
typedef __attribute__((ext_vector_type(8)))  float           v8f;
typedef __attribute__((ext_vector_type(16))) unsigned short  v16us;
typedef __attribute__((ext_vector_type(8)))  unsigned short  v8us;

// Problem constants (fixed by the reference)
#define BATCH   16
#define NTOK    4096
#define CDIM    768
#define HEADS   12
#define HDIM    64
#define GRID    64           // Hg == Wg
#define QKVCOLS 2304         // 3*C
#define SCALE   0.125f       // HDIM^-0.5

__device__ __forceinline__ unsigned short f2bf(float f) {
    unsigned int u = __float_as_uint(f);
    unsigned int r = u + 0x7FFFu + ((u >> 16) & 1u);   // round-to-nearest-even
    return (unsigned short)(r >> 16);
}
__device__ __forceinline__ float bf2f(unsigned short s) {
    return __uint_as_float(((unsigned int)s) << 16);
}

__device__ __forceinline__ v16bf combine_frag(v8us lo, v8us hi) {
    v16us u;
#pragma unroll
    for (int i = 0; i < 8; ++i) { u[i] = lo[i]; u[8 + i] = hi[i]; }
    return __builtin_bit_cast(v16bf, u);
}

// Load a 16x32 bf16 tile fragment (A operand, or B operand holding rows of B^T)
// from GLOBAL memory. base -> (row 0, k 0); ldr = element stride between rows.
// Per-lane: row = lane&15, koff = (lane>>4)*8; halves 0..7 = K[koff..koff+7],
// halves 8..15 = K[koff+16..koff+23]  (matches ISA 16-bit A layout).
__device__ __forceinline__ v16bf load_frag(const unsigned short* __restrict__ base,
                                           int ldr, int lane) {
    int row  = lane & 15;
    int koff = (lane >> 4) << 3;
    const unsigned short* p = base + (size_t)row * (size_t)ldr + koff;
    return combine_frag(*(const v8us*)p, *(const v8us*)(p + 16));
}

// Same fragment gather from an LDS-resident bf16 tile (vectorized ds_load_b128).
__device__ __forceinline__ v16bf load_frag_lds(const unsigned short* base,
                                               int ldr, int lane) {
    int row  = lane & 15;
    int koff = (lane >> 4) << 3;
    const unsigned short* p = base + row * ldr + koff;
    return combine_frag(*(const v8us*)p, *(const v8us*)(p + 16));
}

__device__ __forceinline__ v8f wmma_bf16(v16bf a, v16bf b, v8f c) {
    return __builtin_amdgcn_wmma_f32_16x16x32_bf16(
        /*neg_a=*/false, a, /*neg_b=*/false, b,
        /*c_mod=*/(short)0, c, /*reuse_a=*/false, /*reuse_b=*/false);
}

// ---------------------------------------------------------------------------
// fp32 -> bf16 conversion (8 elements / thread, n divisible by 2048)
// ---------------------------------------------------------------------------
__global__ void __launch_bounds__(256)
cvt_f32_bf16(const float* __restrict__ src, unsigned short* __restrict__ dst, int n) {
    int i = (blockIdx.x * 256 + threadIdx.x) * 8;
    if (i + 8 > n) return;
    float4 a = *(const float4*)(src + i);
    float4 b = *(const float4*)(src + i + 4);
    v8us o;
    o[0] = f2bf(a.x); o[1] = f2bf(a.y); o[2] = f2bf(a.z); o[3] = f2bf(a.w);
    o[4] = f2bf(b.x); o[5] = f2bf(b.y); o[6] = f2bf(b.z); o[7] = f2bf(b.w);
    *(v8us*)(dst + i) = o;
}

// ---------------------------------------------------------------------------
// GEMM: OUT(M,N) = A(M,K) @ Bm(N,K)^T [+ bias], A/Bm bf16, accum fp32.
// Block = 256 threads = 8 waves; block tile 128(M) x 128(N); wave tile 16x128.
// No LDS staging: weights + activations are L2-resident (192 MB L2).
// OUTMODE 0: bf16 output, no bias.  OUTMODE 1: fp32 output + bias.
// ---------------------------------------------------------------------------
template <int OUTMODE>
__global__ void __launch_bounds__(256)
gemm_wmma(const unsigned short* __restrict__ A,
          const unsigned short* __restrict__ Bm,
          const float* __restrict__ bias,
          void* __restrict__ out, int N, int K) {
    const int wave = threadIdx.x >> 5;
    const int lane = threadIdx.x & 31;
    const int m0 = blockIdx.y * 128 + wave * 16;
    const int n0 = blockIdx.x * 128;

    const unsigned short* aBase = A + (size_t)m0 * (size_t)K;

    v8f acc[8];
#pragma unroll
    for (int t = 0; t < 8; ++t) acc[t] = (v8f){0.f, 0.f, 0.f, 0.f, 0.f, 0.f, 0.f, 0.f};

    for (int k = 0; k < K; k += 32) {
        if (k + 32 < K)  // hint next A tile into cache: global_prefetch_b8
            __builtin_prefetch(aBase + (size_t)(lane & 15) * K + k + 32, 0, 3);
        v16bf a = load_frag(aBase + k, K, lane);
#pragma unroll
        for (int t = 0; t < 8; ++t) {
            v16bf b = load_frag(Bm + (size_t)(n0 + t * 16) * (size_t)K + k, K, lane);
            acc[t] = wmma_bf16(a, b, acc[t]);
        }
    }

    const int colL  = lane & 15;
    const int rbase = (lane >> 4) << 3;
#pragma unroll
    for (int t = 0; t < 8; ++t) {
        int gn = n0 + t * 16 + colL;
        float bv = (OUTMODE == 1) ? bias[gn] : 0.f;
#pragma unroll
        for (int r = 0; r < 8; ++r) {
            size_t idx = (size_t)(m0 + rbase + r) * (size_t)N + gn;
            if (OUTMODE == 0) ((unsigned short*)out)[idx] = f2bf(acc[t][r]);
            else              ((float*)out)[idx] = acc[t][r] + bv;
        }
    }
}

// ---------------------------------------------------------------------------
// Axial attention for one (b, head, line): S = Q K^T * SCALE, softmax,
// weights (+)= P, Y (+)= P @ V.  dir 0 = rows (write), dir 1 = cols (add).
// Block = 128 threads = 4 waves; each wave owns a 16-row strip.
// LDS: fp32 scores sS, V^T tile sVT (bf16), probabilities sP (bf16).
// ---------------------------------------------------------------------------
#define S_LD 65      // fp32 score tile stride (padded)
#define V_LD 72      // bf16 V^T tile stride (padded, keeps 16B alignment)
#define P_LD 72      // bf16 P tile stride   (padded, keeps 16B alignment)

__global__ void __launch_bounds__(128)
attn_axial(const unsigned short* __restrict__ qkvb,
           unsigned short* __restrict__ yb,
           float* __restrict__ weights, int dir) {
    __shared__ float sS[GRID * S_LD];
    __shared__ __align__(16) unsigned short sVT[HDIM * V_LD];  // sVT[d][token]
    __shared__ __align__(16) unsigned short sP[GRID * P_LD];   // bf16 probs

    const int job  = blockIdx.x;           // 0 .. B*HEADS*GRID-1
    const int line = job & 63;
    const int hh   = (job >> 6) % HEADS;
    const int b    = job / (GRID * HEADS);

    const int n0        = (dir == 0) ? line * GRID : line;
    const int rowStride = (dir == 0) ? QKVCOLS : QKVCOLS * GRID;

    const unsigned short* qb = qkvb + ((size_t)(b * NTOK + n0)) * QKVCOLS + hh * HDIM;
    const unsigned short* kb = qb + CDIM;
    const unsigned short* vb = qb + 2 * CDIM;

    const int tid  = threadIdx.x;
    const int wave = tid >> 5;
    const int lane = tid & 31;

    // ---- stage V^T tile into LDS (coalesced global reads, transposed store) ----
#pragma unroll
    for (int j = 0; j < 4; ++j) {
        int vec = tid + j * 128;           // 512 x 8-element vectors
        int row = vec >> 3;                // token index
        int cv  = (vec & 7) * 8;           // d offset
        v8us vv = *(const v8us*)(vb + (size_t)row * (size_t)rowStride + cv);
#pragma unroll
        for (int e = 0; e < 8; ++e) sVT[(cv + e) * V_LD + row] = vv[e];
    }

    // ---- scores: wave computes 16x64 strip of S = Q K^T ----
    const int m0 = wave * 16;
    v16bf a0 = load_frag(qb + (size_t)m0 * rowStride,      rowStride, lane);
    v16bf a1 = load_frag(qb + (size_t)m0 * rowStride + 32, rowStride, lane);
#pragma unroll
    for (int t = 0; t < 4; ++t) {
        v16bf b0 = load_frag(kb + (size_t)(t * 16) * rowStride,      rowStride, lane);
        v16bf b1 = load_frag(kb + (size_t)(t * 16) * rowStride + 32, rowStride, lane);
        v8f c = (v8f){0.f, 0.f, 0.f, 0.f, 0.f, 0.f, 0.f, 0.f};
        c = wmma_bf16(a0, b0, c);
        c = wmma_bf16(a1, b1, c);
        int col   = lane & 15;
        int rbase = (lane >> 4) << 3;
#pragma unroll
        for (int r = 0; r < 8; ++r)
            sS[(m0 + rbase + r) * S_LD + t * 16 + col] = c[r] * SCALE;
    }
    __syncthreads();

    // ---- softmax per row: emit weights (fp32, global) + P (bf16, LDS) ----
    if (tid < GRID) {
        float* srow = sS + tid * S_LD;
        float mx = -3.4e38f;
#pragma unroll 8
        for (int j = 0; j < GRID; ++j) mx = fmaxf(mx, srow[j]);
        float sum = 0.f;
#pragma unroll 8
        for (int j = 0; j < GRID; ++j) { float e = __expf(srow[j] - mx); srow[j] = e; sum += e; }
        float inv = 1.f / sum;
        float* wout = weights + ((size_t)(b * HEADS + hh) * GRID + line) * (GRID * GRID)
                              + (size_t)tid * GRID;
        unsigned short* prow = sP + tid * P_LD;
#pragma unroll 8
        for (int j = 0; j < GRID; ++j) {
            float p = srow[j] * inv;
            prow[j] = f2bf(p);
            if (dir == 0) wout[j] = p; else wout[j] += p;
        }
    }
    __syncthreads();

    // ---- O = P @ V : wave computes rows m0..m0+15, all 64 dims ----
    // A fragments (P) hoisted: depend only on k-step. B fragments are rows of
    // V^T in LDS (= columns of V), vectorized ds_load_b128.
    v16bf aF[2];
#pragma unroll
    for (int ks = 0; ks < 2; ++ks)
        aF[ks] = load_frag_lds(sP + m0 * P_LD + ks * 32, P_LD, lane);

    const size_t ybase = (size_t)b * ((size_t)NTOK * CDIM)
                       + (size_t)(hh * GRID + line) * (GRID * HDIM);
#pragma unroll
    for (int t = 0; t < 4; ++t) {
        v8f c = (v8f){0.f, 0.f, 0.f, 0.f, 0.f, 0.f, 0.f, 0.f};
#pragma unroll
        for (int ks = 0; ks < 2; ++ks) {
            v16bf bF = load_frag_lds(sVT + (t * 16) * V_LD + ks * 32, V_LD, lane);
            c = wmma_bf16(aF[ks], bF, c);
        }
        int col   = lane & 15;
        int rbase = (lane >> 4) << 3;
#pragma unroll
        for (int r = 0; r < 8; ++r) {
            size_t idx = ybase + (size_t)(m0 + rbase + r) * HDIM + t * 16 + col;
            if (dir == 0) yb[idx] = f2bf(c[r]);
            else          yb[idx] = f2bf(bf2f(yb[idx]) + c[r]);
        }
    }
}

// ---------------------------------------------------------------------------
// Launch: inputs are {x, qkv_w, proj_w, proj_b}, all fp32.
// d_out = [out (B*N*C fp32) | weights (B*h*64*64*64 fp32)]
// ---------------------------------------------------------------------------
extern "C" void kernel_launch(void* const* d_in, const int* in_sizes, int n_in,
                              void* d_out, int out_size, void* d_ws, size_t ws_size,
                              hipStream_t stream) {
    (void)in_sizes; (void)n_in; (void)out_size; (void)ws_size;

    const float* x      = (const float*)d_in[0];
    const float* qkv_w  = (const float*)d_in[1];
    const float* proj_w = (const float*)d_in[2];
    const float* proj_b = (const float*)d_in[3];

    float* out     = (float*)d_out;
    float* weights = out + (size_t)BATCH * NTOK * CDIM;   // 50,331,648 floats in

    char* ws = (char*)d_ws;
    unsigned short* xb    = (unsigned short*)(ws);                 //  96.0 MB
    unsigned short* wq    = (unsigned short*)(ws + 100663296ull);  //   3.4 MB
    unsigned short* wp    = (unsigned short*)(ws + 104202240ull);  //   1.1 MB
    unsigned short* qkvb  = (unsigned short*)(ws + 105381888ull);  // 288.0 MB
    unsigned short* yb    = (unsigned short*)(ws + 407371776ull);  //  96.0 MB
                                                                   // total ~485 MB

    // 1) fp32 -> bf16 conversions
    cvt_f32_bf16<<<24576, 256, 0, stream>>>(x,      xb, BATCH * NTOK * CDIM);
    cvt_f32_bf16<<<864,   256, 0, stream>>>(qkv_w,  wq, QKVCOLS * CDIM);
    cvt_f32_bf16<<<288,   256, 0, stream>>>(proj_w, wp, CDIM * CDIM);

    // 2) QKV projection: qkvb = xb @ wq^T   (65536 x 2304, K=768), bf16 out
    gemm_wmma<0><<<dim3(QKVCOLS / 128, (BATCH * NTOK) / 128), 256, 0, stream>>>(
        xb, wq, nullptr, qkvb, QKVCOLS, CDIM);

    // 3) axial attention: horizontal (write) then vertical (accumulate)
    attn_axial<<<BATCH * HEADS * GRID, 128, 0, stream>>>(qkvb, yb, weights, 0);
    attn_axial<<<BATCH * HEADS * GRID, 128, 0, stream>>>(qkvb, yb, weights, 1);

    // 4) output projection: out = yb @ wp^T + proj_b  (65536 x 768, K=768), fp32 out
    gemm_wmma<1><<<dim3(CDIM / 128, (BATCH * NTOK) / 128), 256, 0, stream>>>(
        yb, wp, proj_b, out, CDIM, CDIM);
}
